// MixtureOfExperts_37065567764969
// MI455X (gfx1250) — compile-verified, matrix operand order checked by reading, not run on hardware
//
#include <hip/hip_runtime.h>
#include <hip/hip_bf16.h>
#include <math.h>

#define D_MODEL   1024
#define FF_DIM    4096
#define NUM_E     8
#define T_TOKENS  8192
#define MT        128
#define NT        64
#define KT        32
#define MAX_TILES 72          // T/MT + NUM_E (worst-case ceil split)
#define LDST      40          // LDS row stride in halfs (bank spread)

typedef __attribute__((ext_vector_type(16))) __bf16 v16bf;
typedef __attribute__((ext_vector_type(8)))  float  v8f;

union BF16x16 { v16bf v; unsigned int u[8]; };

__device__ __forceinline__ unsigned int f2bf(float f) {
  unsigned int u = __float_as_uint(f);
  return (u + 0x7fffu + ((u >> 16) & 1u)) >> 16;   // RNE
}
__device__ __forceinline__ unsigned int pack2bf(float a, float b) {
  return f2bf(a) | (f2bf(b) << 16);
}

// memory -> LDS direct (ASYNCcnt-tracked), GVS addressing: saddr base + vaddr byte offset
__device__ __forceinline__ void async_b128_to_lds(void* ldsdst, const void* sbase,
                                                  unsigned int voff_bytes) {
  unsigned int ldsaddr = (unsigned int)(unsigned long long)ldsdst;  // low 32 = LDS addr
  asm volatile("global_load_async_to_lds_b128 %0, %1, %2"
               :: "v"(ldsaddr), "v"(voff_bytes), "s"(sbase) : "memory");
}
__device__ __forceinline__ void wait_async0() {
  asm volatile("s_wait_asynccnt 0x0" ::: "memory");
}

// ---------------- workspace layout (bytes) ----------------
#define OFF_XB    ((size_t)0)                                   // T*D bf16
#define OFF_H     ((size_t)16777216)                            // (T+MT)*F bf16
#define OFF_COEF  ((size_t)(OFF_H + (size_t)(T_TOKENS+MT)*FF_DIM*2))
#define OFF_LISTS ((size_t)(OFF_COEF + 2ull*T_TOKENS*4))
#define OFF_CNT   ((size_t)(OFF_LISTS + 2ull*NUM_E*T_TOKENS*4))
#define OFF_DESC  ((size_t)(OFF_CNT + 64))

// ---------------- tiny kernels ----------------
__global__ void moe_init(int* counts) {
  if (threadIdx.x < 16) counts[threadIdx.x] = 0;
}

__global__ void moe_xcast(const float* __restrict__ x, unsigned short* __restrict__ xb) {
  size_t i = ((size_t)blockIdx.x * blockDim.x + threadIdx.x) * 4;
  float4 v = *(const float4*)(x + i);
  *(uint2*)(xb + i) = make_uint2(pack2bf(v.x, v.y), pack2bf(v.z, v.w));
}

// one wave32 per token: logits -> softmax -> top2 -> append to (slot,expert) list
__global__ void moe_router(const float* __restrict__ x, const float* __restrict__ rw,
                           const float* __restrict__ rb, int* __restrict__ counts,
                           int* __restrict__ lists, float* __restrict__ coefs) {
  int lane = threadIdx.x & 31;
  int t = blockIdx.x * 8 + (threadIdx.x >> 5);
  float acc[NUM_E];
#pragma unroll
  for (int e = 0; e < NUM_E; e++) acc[e] = 0.f;
  const float* xr = x + (size_t)t * D_MODEL;
  for (int d = lane; d < D_MODEL; d += 32) {
    float xv = xr[d];
#pragma unroll
    for (int e = 0; e < NUM_E; e++) acc[e] += xv * rw[e * D_MODEL + d];
  }
#pragma unroll
  for (int e = 0; e < NUM_E; e++) {
    float v = acc[e];
    for (int off = 16; off >= 1; off >>= 1) v += __shfl_xor(v, off, 32);
    acc[e] = v;
  }
  if (lane == 0) {
    float l[NUM_E];
#pragma unroll
    for (int e = 0; e < NUM_E; e++) l[e] = acc[e] + rb[e];
    int e0 = 0;
    for (int e = 1; e < NUM_E; e++) if (l[e] > l[e0]) e0 = e;
    int e1 = (e0 == 0) ? 1 : 0;
    for (int e = 0; e < NUM_E; e++) if (e != e0 && l[e] > l[e1]) e1 = e;
    float w0 = 1.f / (1.f + expf(l[e1] - l[e0]));   // renormalized top-2
    int p0 = atomicAdd(&counts[e0], 1);
    lists[e0 * T_TOKENS + p0] = t;
    coefs[t] = w0;
    int p1 = atomicAdd(&counts[NUM_E + e1], 1);
    lists[(NUM_E + e1) * T_TOKENS + p1] = t;
    coefs[T_TOKENS + t] = 1.f - w0;
  }
}

// build fixed-size tile descriptor tables: (expert, m_start, row_base, count)
__global__ void moe_scan(const int* __restrict__ counts, int4* __restrict__ descs) {
  if (threadIdx.x != 0) return;
  for (int s = 0; s < 2; s++) {
    int base = 0, idx = 0;
    for (int e = 0; e < NUM_E; e++) {
      int c = counts[s * NUM_E + e];
      int nt = (c + MT - 1) / MT;
      for (int i = 0; i < nt && idx < MAX_TILES; i++)
        descs[s * MAX_TILES + idx++] = make_int4(e, i * MT, base, c);
      base += c;
    }
    for (; idx < MAX_TILES; idx++) descs[s * MAX_TILES + idx] = make_int4(-1, 0, 0, 0);
  }
}

// ---------------- GEMM1: H = gelu(X_e @ w1_e^T + b1_e), bf16 out ----------------
__global__ __launch_bounds__(256) void moe_ffn1(
    const unsigned short* __restrict__ xb, const float* __restrict__ w1,
    const float* __restrict__ b1, const int* __restrict__ lists,
    const int4* __restrict__ descs, unsigned short* __restrict__ H, int slot) {
  __shared__ unsigned short Al[MT * LDST];
  __shared__ unsigned short Bl[NT * LDST];
  __shared__ int trow[MT];

  int4 d = descs[slot * MAX_TILES + blockIdx.x];
  int e = d.x;
  if (e < 0) return;
  int mstart = d.y, base = d.z, cnt = d.w;
  int tid = threadIdx.x, ny = blockIdx.y;
  int lane = tid & 31, wave = tid >> 5, half = lane >> 4, mn = lane & 15;
  const int* lst = lists + (slot * NUM_E + e) * T_TOKENS;
  if (tid < MT) {
    int p = mstart + tid; if (p > cnt - 1) p = cnt - 1;
    trow[tid] = lst[p];
  }
  const float* wexp = w1 + (size_t)e * FF_DIM * D_MODEL;
  int lrow = tid >> 2, c8 = (tid & 3) * 8;

  v8f acc[4];
  v8f zero = {0.f, 0.f, 0.f, 0.f, 0.f, 0.f, 0.f, 0.f};
#pragma unroll
  for (int j = 0; j < 4; j++) acc[j] = zero;

  __syncthreads();
  unsigned int tokA = (unsigned int)trow[lrow];        // rows this thread feeds
  unsigned int tokB = (unsigned int)trow[lrow + 64];

  for (int kk = 0; kk < D_MODEL; kk += KT) {
    // A: 128x32 bf16, gathered token rows, direct global->LDS async copy
    async_b128_to_lds(&Al[lrow * LDST + c8], xb,
                      (tokA * D_MODEL + kk + c8) * 2u);
    async_b128_to_lds(&Al[(lrow + 64) * LDST + c8], xb,
                      (tokB * D_MODEL + kk + c8) * 2u);
    {                                                   // B: 64x32 fp32 -> bf16
      const float* srcf = wexp + (size_t)(ny * NT + lrow) * D_MODEL + kk + c8;
      float4 f0 = ((const float4*)srcf)[0], f1 = ((const float4*)srcf)[1];
      if (kk + KT < D_MODEL) __builtin_prefetch(srcf + KT, 0, 3);
      unsigned int* dst = (unsigned int*)&Bl[lrow * LDST + c8];
      dst[0] = pack2bf(f0.x, f0.y); dst[1] = pack2bf(f0.z, f0.w);
      dst[2] = pack2bf(f1.x, f1.y); dst[3] = pack2bf(f1.z, f1.w);
    }
    wait_async0();
    __syncthreads();
    BF16x16 a;
    int arow = wave * 16 + mn;
#pragma unroll
    for (int v = 0; v < 4; v++) a.u[v]     = *(const unsigned int*)&Al[arow * LDST + 8 * half + 2 * v];
#pragma unroll
    for (int v = 0; v < 4; v++) a.u[4 + v] = *(const unsigned int*)&Al[arow * LDST + 16 + 8 * half + 2 * v];
#pragma unroll
    for (int j = 0; j < 4; j++) {
      BF16x16 b;
      int brow = j * 16 + mn;
#pragma unroll
      for (int v = 0; v < 4; v++) b.u[v]     = *(const unsigned int*)&Bl[brow * LDST + 8 * half + 2 * v];
#pragma unroll
      for (int v = 0; v < 4; v++) b.u[4 + v] = *(const unsigned int*)&Bl[brow * LDST + 16 + 8 * half + 2 * v];
      acc[j] = __builtin_amdgcn_wmma_f32_16x16x32_bf16(false, a.v, false, b.v,
                                                       (short)0, acc[j], false, false);
    }
    __syncthreads();
  }

  const float* b1e = b1 + e * FF_DIM;
#pragma unroll
  for (int j = 0; j < 4; j++) {
    int ncol = ny * NT + j * 16 + mn;
    float bias = b1e[ncol];
#pragma unroll
    for (int r = 0; r < 8; r++) {
      int mloc = wave * 16 + half * 8 + r;
      if (mstart + mloc < cnt) {
        float v = acc[j][r] + bias;
        float g = 0.5f * v * (1.0f + erff(v * 0.70710678118654752f));   // exact GELU
        H[(size_t)(base + mstart + mloc) * FF_DIM + ncol] = (unsigned short)f2bf(g);
      }
    }
  }
}

// ---------------- GEMM2: out[t] (=|+=) coef * (H @ w2_e^T + b2_e) ----------------
__global__ __launch_bounds__(256) void moe_ffn2(
    const unsigned short* __restrict__ H, const float* __restrict__ w2,
    const float* __restrict__ b2, const int* __restrict__ lists,
    const float* __restrict__ coefs, const int4* __restrict__ descs,
    float* __restrict__ out, int slot, int accum) {
  __shared__ unsigned short Al[MT * LDST];
  __shared__ unsigned short Bl[NT * LDST];
  __shared__ int trow[MT];

  int4 d = descs[slot * MAX_TILES + blockIdx.x];
  int e = d.x;
  if (e < 0) return;
  int mstart = d.y, base = d.z, cnt = d.w;
  int tid = threadIdx.x, ny = blockIdx.y;
  int lane = tid & 31, wave = tid >> 5, half = lane >> 4, mn = lane & 15;
  const int* lst = lists + (slot * NUM_E + e) * T_TOKENS;
  if (tid < MT) {
    int p = mstart + tid; if (p > cnt - 1) p = cnt - 1;
    trow[tid] = lst[p];
  }
  const float* wexp = w2 + (size_t)e * D_MODEL * FF_DIM;
  int lrow = tid >> 2, c8 = (tid & 3) * 8;
  unsigned int hr0 = (unsigned int)(base + mstart + lrow);        // contiguous H rows
  unsigned int hr1 = hr0 + 64u;

  v8f acc[4];
  v8f zero = {0.f, 0.f, 0.f, 0.f, 0.f, 0.f, 0.f, 0.f};
#pragma unroll
  for (int j = 0; j < 4; j++) acc[j] = zero;

  __syncthreads();
  for (int kk = 0; kk < FF_DIM; kk += KT) {
    // A: 128x32 bf16 from H, direct global->LDS async copy
    async_b128_to_lds(&Al[lrow * LDST + c8], H,
                      (hr0 * FF_DIM + kk + c8) * 2u);
    async_b128_to_lds(&Al[(lrow + 64) * LDST + c8], H,
                      (hr1 * FF_DIM + kk + c8) * 2u);
    {                                                   // B: 64x32 fp32 -> bf16
      const float* srcf = wexp + (size_t)(ny * NT + lrow) * FF_DIM + kk + c8;
      float4 f0 = ((const float4*)srcf)[0], f1 = ((const float4*)srcf)[1];
      if (kk + KT < FF_DIM) __builtin_prefetch(srcf + KT, 0, 3);
      unsigned int* dst = (unsigned int*)&Bl[lrow * LDST + c8];
      dst[0] = pack2bf(f0.x, f0.y); dst[1] = pack2bf(f0.z, f0.w);
      dst[2] = pack2bf(f1.x, f1.y); dst[3] = pack2bf(f1.z, f1.w);
    }
    wait_async0();
    __syncthreads();
    BF16x16 a;
    int arow = wave * 16 + mn;
#pragma unroll
    for (int v = 0; v < 4; v++) a.u[v]     = *(const unsigned int*)&Al[arow * LDST + 8 * half + 2 * v];
#pragma unroll
    for (int v = 0; v < 4; v++) a.u[4 + v] = *(const unsigned int*)&Al[arow * LDST + 16 + 8 * half + 2 * v];
#pragma unroll
    for (int j = 0; j < 4; j++) {
      BF16x16 b;
      int brow = j * 16 + mn;
#pragma unroll
      for (int v = 0; v < 4; v++) b.u[v]     = *(const unsigned int*)&Bl[brow * LDST + 8 * half + 2 * v];
#pragma unroll
      for (int v = 0; v < 4; v++) b.u[4 + v] = *(const unsigned int*)&Bl[brow * LDST + 16 + 8 * half + 2 * v];
      acc[j] = __builtin_amdgcn_wmma_f32_16x16x32_bf16(false, a.v, false, b.v,
                                                       (short)0, acc[j], false, false);
    }
    __syncthreads();
  }

  const float* b2e = b2 + e * D_MODEL;
#pragma unroll
  for (int j = 0; j < 4; j++) {
    int ncol = ny * NT + j * 16 + mn;
    float bias = b2e[ncol];
#pragma unroll
    for (int r = 0; r < 8; r++) {
      int mloc = wave * 16 + half * 8 + r;
      if (mstart + mloc < cnt) {
        int t = trow[mloc];
        float cf = coefs[slot * T_TOKENS + t];
        float val = cf * (acc[j][r] + bias);
        size_t oi = (size_t)t * D_MODEL + ncol;
        if (accum) out[oi] += val; else out[oi] = val;
      }
    }
  }
}

// ---------------- launch ----------------
extern "C" void kernel_launch(void* const* d_in, const int* in_sizes, int n_in,
                              void* d_out, int out_size, void* d_ws, size_t ws_size,
                              hipStream_t stream) {
  const float* x  = (const float*)d_in[0];
  const float* rw = (const float*)d_in[1];
  const float* rb = (const float*)d_in[2];
  const float* w1 = (const float*)d_in[3];
  const float* b1 = (const float*)d_in[4];
  const float* w2 = (const float*)d_in[5];
  const float* b2 = (const float*)d_in[6];
  float* out = (float*)d_out;
  char* ws = (char*)d_ws;

  unsigned short* xb   = (unsigned short*)(ws + OFF_XB);
  unsigned short* Hbuf = (unsigned short*)(ws + OFF_H);
  float* coefs = (float*)(ws + OFF_COEF);
  int*   lists = (int*)(ws + OFF_LISTS);
  int*   cnts  = (int*)(ws + OFF_CNT);
  int4*  descs = (int4*)(ws + OFF_DESC);

  moe_init<<<1, 32, 0, stream>>>(cnts);
  moe_xcast<<<(T_TOKENS * D_MODEL) / 1024, 256, 0, stream>>>(x, xb);
  moe_router<<<T_TOKENS / 8, 256, 0, stream>>>(x, rw, rb, cnts, lists, coefs);
  moe_scan<<<1, 32, 0, stream>>>(cnts, descs);
  for (int s = 0; s < 2; s++) {
    moe_ffn1<<<dim3(MAX_TILES, FF_DIM / NT), 256, 0, stream>>>(xb, w1, b1, lists, descs, Hbuf, s);
    moe_ffn2<<<dim3(MAX_TILES, D_MODEL / NT), 256, 0, stream>>>(Hbuf, w2, b2, lists, coefs, descs, out, s, s);
  }
}
